// Block_27994596835704
// MI455X (gfx1250) — compile-verified, matrix operand order checked by reading, not run on hardware
//
#include <hip/hip_runtime.h>
#include <hip/hip_bf16.h>

typedef __attribute__((ext_vector_type(16))) _Float16 v16h;
typedef __attribute__((ext_vector_type(8)))  float    v8f;

#define DH 128

// ---------------------------------------------------------------------------
// Zero-fill (agg buffers must be zeroed every launch for determinism)
// ---------------------------------------------------------------------------
__global__ __launch_bounds__(256)
void zero_f32(float* __restrict__ p, long long n4) {
    long long i = (long long)blockIdx.x * blockDim.x + threadIdx.x;
    if (i < n4) ((float4*)p)[i] = make_float4(0.f, 0.f, 0.f, 0.f);
}

// ---------------------------------------------------------------------------
// Pack one 128x128 fp32 weight matrix into WMMA B-fragment layout (f16).
// Layout per (kt, nt): 32 lanes x 16 halves. For lane group g = lane>>4:
//   halves h=0..7  -> K = kt*32 + g*8 + h
//   halves h=8..15 -> K = kt*32 + g*8 + 16 + (h-8)
//   N = nt*16 + (lane & 15)
// (CDNA5 ISA 7.12.2, 16-bit B 32x16 striping)
// ---------------------------------------------------------------------------
__global__ __launch_bounds__(256)
void pack_weights(const float* __restrict__ W, _Float16* __restrict__ Wp) {
    int idx = blockIdx.x * blockDim.x + threadIdx.x;   // 4*8*32*16 = 16384
    if (idx >= 4 * 8 * 32 * 16) return;
    int h    = idx & 15;
    int lane = (idx >> 4) & 31;
    int nt   = (idx >> 9) & 7;
    int kt   = (idx >> 12) & 3;
    int grp  = lane >> 4;
    int k    = kt * 32 + grp * 8 + ((h < 8) ? h : (16 + (h - 8)));
    int n    = nt * 16 + (lane & 15);
    Wp[idx]  = (_Float16)W[k * DH + n];
}

// ---------------------------------------------------------------------------
// Edge aggregation: agg[dst] += src_rows[src], one wave per edge.
// Indices broadcast to SGPRs via readfirstlane (wave-uniform).
// ---------------------------------------------------------------------------
__global__ __launch_bounds__(256)
void edge_aggregate(const float* __restrict__ srcRows,
                    const int* __restrict__ esrc, const int* __restrict__ edst,
                    float* __restrict__ agg, int nEdges) {
    int edge = blockIdx.x * (blockDim.x >> 5) + (threadIdx.x >> 5);
    if (edge >= nEdges) return;
    int lane = threadIdx.x & 31;
    int s = __builtin_amdgcn_readfirstlane(esrc[edge]);
    int d = __builtin_amdgcn_readfirstlane(edst[edge]);
    float4 v = ((const float4*)(srcRows + (long long)s * DH))[lane];
    float* out = agg + (long long)d * DH + (lane << 2);
    atomicAdd(out + 0, v.x);
    atomicAdd(out + 1, v.y);
    atomicAdd(out + 2, v.z);
    atomicAdd(out + 3, v.w);
}

// ---------------------------------------------------------------------------
// WMMA GEMM: Out[N,128] = f( (A0 (+A1)) @ W + bias ) (+Res)
//   Accumulators start from WMMA's inline-0 C operand; bias is fused into the
//   LDS-readback epilogue (branch-free, coalesced b128 stores).
// ---------------------------------------------------------------------------
template <bool HAS_A1, bool HAS_RES, bool RELU>
__global__ __launch_bounds__(128)
void gemm128_wmma(const float* __restrict__ A0,
                  const float* __restrict__ A1,     // addend (HAS_A1)
                  const _Float16* __restrict__ Wp,  // [4][8][32][16] halves
                  const float* __restrict__ bias,   // [128]
                  const float* __restrict__ Res,    // residual (HAS_RES)
                  float* __restrict__ Out,
                  int nRowTiles) {
    __shared__ float lds[4][16 * DH];                // 8KB per wave, 32KB/block
    int wave = blockIdx.x * (blockDim.x >> 5) + (threadIdx.x >> 5);
    if (wave >= nRowTiles) return;                   // wave-uniform: EXEC all-1
    int lane = threadIdx.x & 31;
    int grp  = lane >> 4;
    int col  = lane & 15;
    float* myLds = lds[threadIdx.x >> 5];
    long long row = (long long)wave * 16 + col;      // A layout: lane&15 = M

    // Build A fragments (16-bit A 16x32 layout): per lane group g,
    // halves 0..7 <- K = kt*32 + g*8 .. +7 ; halves 8..15 <- +16 offset.
    const float* ar0 = A0 + row * DH;
    const float* ar1 = HAS_A1 ? (A1 + row * DH) : nullptr;
    v16h afrag[4];
#pragma unroll
    for (int kt = 0; kt < 4; ++kt) {
        int off = kt * 32 + grp * 8;
        float4 q0 = *(const float4*)(ar0 + off);
        float4 q1 = *(const float4*)(ar0 + off + 4);
        float4 q2 = *(const float4*)(ar0 + off + 16);
        float4 q3 = *(const float4*)(ar0 + off + 20);
        if (HAS_A1) {
            float4 b;
            b = *(const float4*)(ar1 + off);      q0.x += b.x; q0.y += b.y; q0.z += b.z; q0.w += b.w;
            b = *(const float4*)(ar1 + off + 4);  q1.x += b.x; q1.y += b.y; q1.z += b.z; q1.w += b.w;
            b = *(const float4*)(ar1 + off + 16); q2.x += b.x; q2.y += b.y; q2.z += b.z; q2.w += b.w;
            b = *(const float4*)(ar1 + off + 20); q3.x += b.x; q3.y += b.y; q3.z += b.z; q3.w += b.w;
        }
        v16h a;
        a[0]  = (_Float16)q0.x; a[1]  = (_Float16)q0.y; a[2]  = (_Float16)q0.z; a[3]  = (_Float16)q0.w;
        a[4]  = (_Float16)q1.x; a[5]  = (_Float16)q1.y; a[6]  = (_Float16)q1.z; a[7]  = (_Float16)q1.w;
        a[8]  = (_Float16)q2.x; a[9]  = (_Float16)q2.y; a[10] = (_Float16)q2.z; a[11] = (_Float16)q2.w;
        a[12] = (_Float16)q3.x; a[13] = (_Float16)q3.y; a[14] = (_Float16)q3.z; a[15] = (_Float16)q3.w;
        afrag[kt] = a;
    }

    const v16h* wfrag = (const v16h*)Wp;
#pragma unroll
    for (int nt = 0; nt < 8; ++nt) {
        v8f acc = {};                               // inline-0 C for first WMMA
#pragma unroll
        for (int kt = 0; kt < 4; ++kt) {
            v16h b = wfrag[(kt * 8 + nt) * 32 + lane];
            acc = __builtin_amdgcn_wmma_f32_16x16x32_f16(
                false, afrag[kt], false, b, (short)0, acc, false, false);
        }
        // C/D layout: VGPR r -> M = r + grp*8, N = lane&15; park tile in LDS
#pragma unroll
        for (int r = 0; r < 8; ++r)
            myLds[(r + grp * 8) * DH + nt * 16 + col] = acc[r];
    }

    // Coalesced b128 writeback; fused bias + relu + residual, branch-free.
    float4 bvec = ((const float4*)bias)[lane];       // columns lane*4 .. +3
#pragma unroll
    for (int rl = 0; rl < 16; ++rl) {
        float4 v = ((const float4*)myLds)[rl * 32 + lane];
        v.x += bvec.x; v.y += bvec.y; v.z += bvec.z; v.w += bvec.w;
        if (RELU) {
            v.x = fmaxf(v.x, 0.f); v.y = fmaxf(v.y, 0.f);
            v.z = fmaxf(v.z, 0.f); v.w = fmaxf(v.w, 0.f);
        }
        long long m = (long long)wave * 16 + rl;
        if (HAS_RES) {
            float4 rv = ((const float4*)(Res + m * DH))[lane];
            v.x += rv.x; v.y += rv.y; v.z += rv.z; v.w += rv.w;
        }
        ((float4*)(Out + m * DH))[lane] = v;
    }
}

// ---------------------------------------------------------------------------
// t2 = relu(t + ga + gb), elementwise float4
// ---------------------------------------------------------------------------
__global__ __launch_bounds__(256)
void fuse_t2(const float* __restrict__ t, const float* __restrict__ ga,
             const float* __restrict__ gb, float* __restrict__ out, long long n4) {
    long long i = (long long)blockIdx.x * blockDim.x + threadIdx.x;
    if (i >= n4) return;
    float4 a = ((const float4*)t)[i];
    float4 b = ((const float4*)ga)[i];
    float4 c = ((const float4*)gb)[i];
    float4 o;
    o.x = fmaxf(a.x + b.x + c.x, 0.f);
    o.y = fmaxf(a.y + b.y + c.y, 0.f);
    o.z = fmaxf(a.z + b.z + c.z, 0.f);
    o.w = fmaxf(a.w + b.w + c.w, 0.f);
    ((float4*)out)[i] = o;
}

// ---------------------------------------------------------------------------
// t_out = t2 + LayerNorm(u) * g + b    (one wave32 per 128-wide row)
// ---------------------------------------------------------------------------
__global__ __launch_bounds__(256)
void ln_residual(const float* __restrict__ u, const float* __restrict__ t2,
                 const float* __restrict__ g, const float* __restrict__ b,
                 float* __restrict__ out, int nRows) {
    int wave = blockIdx.x * (blockDim.x >> 5) + (threadIdx.x >> 5);
    if (wave >= nRows) return;
    int lane = threadIdx.x & 31;
    long long base = (long long)wave * DH;
    float4 v = ((const float4*)(u + base))[lane];
    float s  = v.x + v.y + v.z + v.w;
    float sq = v.x * v.x + v.y * v.y + v.z * v.z + v.w * v.w;
#pragma unroll
    for (int off = 16; off > 0; off >>= 1) {
        s  += __shfl_xor(s, off, 32);
        sq += __shfl_xor(sq, off, 32);
    }
    float mean = s * (1.0f / 128.0f);
    float var  = sq * (1.0f / 128.0f) - mean * mean;
    float inv  = rsqrtf(var + 1e-5f);
    float4 tv = ((const float4*)(t2 + base))[lane];
    float4 gv = ((const float4*)g)[lane];
    float4 bv = ((const float4*)b)[lane];
    float4 o;
    o.x = tv.x + (v.x - mean) * inv * gv.x + bv.x;
    o.y = tv.y + (v.y - mean) * inv * gv.y + bv.y;
    o.z = tv.z + (v.z - mean) * inv * gv.z + bv.z;
    o.w = tv.w + (v.w - mean) * inv * gv.w + bv.w;
    ((float4*)(out + base))[lane] = o;
}

// ---------------------------------------------------------------------------
// Launch
// ---------------------------------------------------------------------------
extern "C" void kernel_launch(void* const* d_in, const int* in_sizes, int n_in,
                              void* d_out, int out_size, void* d_ws, size_t ws_size,
                              hipStream_t stream) {
    const float* x    = (const float*)d_in[0];
    const float* t    = (const float*)d_in[1];
    const int*   e_t  = (const int*)d_in[2];
    const int*   e_x  = (const int*)d_in[3];
    const float* W1a  = (const float*)d_in[4];
    const float* b1a  = (const float*)d_in[5];
    const float* W2a  = (const float*)d_in[6];
    const float* b2a  = (const float*)d_in[7];
    const float* W1b  = (const float*)d_in[8];
    const float* b1b  = (const float*)d_in[9];
    const float* W2b  = (const float*)d_in[10];
    const float* b2b  = (const float*)d_in[11];
    const float* Wo   = (const float*)d_in[12];
    const float* bo   = (const float*)d_in[13];
    const float* ln_g = (const float*)d_in[14];
    const float* ln_b = (const float*)d_in[15];
    const float* Wf1  = (const float*)d_in[16];
    const float* bf1  = (const float*)d_in[17];
    const float* Wf2  = (const float*)d_in[18];
    const float* bf2  = (const float*)d_in[19];

    const int N = in_sizes[0] / DH;        // 50000
    const int E = in_sizes[2] / 2;         // 1,600,000
    const long long ND = (long long)N * DH;

    float* buf0 = (float*)d_ws;
    float* buf1 = buf0 + ND;
    float* buf2 = buf1 + ND;
    _Float16* wp = (_Float16*)(buf2 + ND);
    const int WPN = 4 * 8 * 32 * 16;       // 16384 halves per matrix
    _Float16* wp0 = wp;            // W1a
    _Float16* wp1 = wp + WPN;      // W2a
    _Float16* wp2 = wp + 2 * WPN;  // W1b
    _Float16* wp3 = wp + 3 * WPN;  // W2b
    _Float16* wp4 = wp + 4 * WPN;  // Wo
    _Float16* wp5 = wp + 5 * WPN;  // Wf1
    _Float16* wp6 = wp + 6 * WPN;  // Wf2

    float* x_out = (float*)d_out;
    float* t_out = x_out + ND;

    const int nRowTiles = N / 16;          // 3125 (exact)
    dim3 gB((nRowTiles + 3) / 4), tB(128); // 4 waves/block

    // 0) zero agg buffers (buf0, buf1 contiguous)
    {
        long long n4 = (2 * ND) / 4;
        zero_f32<<<(unsigned)((n4 + 255) / 256), 256, 0, stream>>>(buf0, n4);
    }
    // 1) pack all weights to f16 fragments
    {
        dim3 g((WPN + 255) / 256), b(256);
        pack_weights<<<g, b, 0, stream>>>(W1a, wp0);
        pack_weights<<<g, b, 0, stream>>>(W2a, wp1);
        pack_weights<<<g, b, 0, stream>>>(W1b, wp2);
        pack_weights<<<g, b, 0, stream>>>(W2b, wp3);
        pack_weights<<<g, b, 0, stream>>>(Wo,  wp4);
        pack_weights<<<g, b, 0, stream>>>(Wf1, wp5);
        pack_weights<<<g, b, 0, stream>>>(Wf2, wp6);
    }
    // 2) edge aggregations (dominant memory phase)
    {
        unsigned blocks = (unsigned)((E + 7) / 8);   // 8 waves per 256-thr block
        edge_aggregate<<<blocks, 256, 0, stream>>>(t, e_t, e_t + E, buf0, E);
        edge_aggregate<<<blocks, 256, 0, stream>>>(x, e_x, e_x + E, buf1, E);
    }
    // 3) GIN MLPs (pre-activation add fused into A-load)
    gemm128_wmma<true,  false, true ><<<gB, tB, 0, stream>>>(t,    buf0,    wp0, b1a, nullptr, buf2, nRowTiles); // h_a
    gemm128_wmma<false, false, false><<<gB, tB, 0, stream>>>(buf2, nullptr, wp1, b2a, nullptr, buf0, nRowTiles); // g_a
    gemm128_wmma<true,  false, true ><<<gB, tB, 0, stream>>>(t,    buf1,    wp2, b1b, nullptr, buf2, nRowTiles); // h_b
    gemm128_wmma<false, false, false><<<gB, tB, 0, stream>>>(buf2, nullptr, wp3, b2b, nullptr, buf1, nRowTiles); // g_b
    // 4) t2 = relu(t + g_a + g_b)
    fuse_t2<<<(unsigned)((ND / 4 + 255) / 256), 256, 0, stream>>>(t, buf0, buf1, buf2, ND / 4);
    // 5) out branch: u = relu(t2@Wo + bo); t_out = t2 + LN(u)
    gemm128_wmma<false, false, true ><<<gB, tB, 0, stream>>>(buf2, nullptr, wp4, bo, nullptr, buf0, nRowTiles);
    {
        unsigned blocks = (unsigned)(((long long)N * 32 + 255) / 256);
        ln_residual<<<blocks, 256, 0, stream>>>(buf0, buf2, ln_g, ln_b, t_out, N);
    }
    // 6) fc_x branch: v = relu(x@Wf1+bf1); x_out = x + v@Wf2 + bf2
    gemm128_wmma<false, false, true ><<<gB, tB, 0, stream>>>(x,    nullptr, wp5, bf1, nullptr, buf1, nRowTiles);
    gemm128_wmma<false, true,  false><<<gB, tB, 0, stream>>>(buf1, nullptr, wp6, bf2, x, x_out, nRowTiles);
}
